// MultiHeadAttention_40888088658344
// MI455X (gfx1250) — compile-verified, matrix operand order checked by reading, not run on hardware
//
#include <hip/hip_runtime.h>
#include <math.h>

// Problem constants (match reference)
#define DM 1024   // d_model
#define NH 16     // heads
#define DF 64     // d_feature
#define BB 4      // batch
#define SS 2048   // seq len

typedef __attribute__((ext_vector_type(16))) _Float16 v16h;
typedef __attribute__((ext_vector_type(8)))  _Float16 v8h;
typedef __attribute__((ext_vector_type(8)))  float    v8f;

// ---------------------------------------------------------------------------
// WMMA fragment loader. CDNA5 16-bit A (16xK) / B (Kx16) fragments:
//   lanes 0-15  : row (lane&15), halves 0..7 = K 0..7,   halves 8..15 = K 16..23
//   lanes 16-31 : row (lane&15), halves 0..7 = K 8..15,  halves 8..15 = K 24..31
// With "row" contiguous in memory (stride in halves), each lane loads two
// contiguous 16-byte chunks. base points at (tile_row0, tile_k0).
// ---------------------------------------------------------------------------
__device__ __forceinline__ v16h ld_frag(const _Float16* base, int stride) {
  const int lane = threadIdx.x & 31;
  const _Float16* p = base + (size_t)(lane & 15) * (size_t)stride + ((lane >> 4) << 3);
  v8h lo = *(const v8h*)(p);
  v8h hi = *(const v8h*)(p + 16);
  v16h r;
#pragma unroll
  for (int i = 0; i < 8; ++i) { r[i] = lo[i]; r[i + 8] = hi[i]; }
  return r;
}

__device__ __forceinline__ v8f wmma_f16(v16h a, v16h b, v8f c) {
  return __builtin_amdgcn_wmma_f32_16x16x32_f16(false, a, false, b, (short)0, c,
                                                false, false);
}

// Reductions within each 16-lane half (D-tile rows span 16 lanes).
__device__ __forceinline__ float redsum16(float x) {
#pragma unroll
  for (int m = 1; m < 16; m <<= 1) x += __shfl_xor(x, m, 32);
  return x;
}
__device__ __forceinline__ float redmax16(float x) {
#pragma unroll
  for (int m = 1; m < 16; m <<= 1) x = fmaxf(x, __shfl_xor(x, m, 32));
  return x;
}

// ---------------------------------------------------------------------------
// Precision / layout staging kernels
// ---------------------------------------------------------------------------
__global__ void k_cvt_sin(const float* __restrict__ x, _Float16* __restrict__ y, int n) {
  int i = blockIdx.x * blockDim.x + threadIdx.x;
  if (i < n) y[i] = (_Float16)x[i];
}

// Wx [H, D, F] f32  ->  WxT [H, F, D] f16  (B-fragment friendly)
__global__ void k_cvt_wxT(const float* __restrict__ wx, _Float16* __restrict__ wxT) {
  int i = blockIdx.x * blockDim.x + threadIdx.x;   // NH*DM*DF threads exactly
  int h = i >> 16;
  int rem = i & 65535;
  int d = rem >> 6;
  int f = rem & 63;
  wxT[(size_t)(h * DF + f) * DM + d] = (_Float16)wx[i];
}

// Wp [in, out] f32 -> WpT [out, in] f16
__global__ void k_cvt_wpT(const float* __restrict__ wp, _Float16* __restrict__ wpT) {
  int i = blockIdx.x * blockDim.x + threadIdx.x;   // DM*DM threads exactly
  int r = i >> 10;        // in
  int c = i & 1023;       // out
  wpT[(size_t)c * DM + r] = (_Float16)wp[i];
}

// ---------------------------------------------------------------------------
// Ping-pong GEMM inner stage: issue loads for k-chunk `dnext` into (a_n,b_n),
// then run the 4 WMMAs on (a_c,b_c). No register copies between stages.
// ---------------------------------------------------------------------------
__device__ __forceinline__ void gemm_stage(const _Float16* abase,
                                           const _Float16* wbase,
                                           int dnext,
                                           v16h& a_c, v16h (&b_c)[4],
                                           v16h& a_n, v16h (&b_n)[4],
                                           v8f (&acc)[4]) {
  a_n = ld_frag(abase + dnext, DM);
#pragma unroll
  for (int nf = 0; nf < 4; ++nf)
    b_n[nf] = ld_frag(wbase + (size_t)(nf * 16) * DM + dnext, DM);
#pragma unroll
  for (int nf = 0; nf < 4; ++nf) acc[nf] = wmma_f16(a_c, b_c[nf], acc[nf]);
}

// ---------------------------------------------------------------------------
// Kernel 1: per-head projection  h = sin @ Wx[h] + bx[h]
// One wave: 16 rows x 64 features of one head; ping-pong pipelined fragments.
// Outputs: hn [B,H,S,F] f16 (rows normalized -> QK^T is cosine similarity),
//          hT [B,H,F,S] f16 (unnormalized, B-operand for P@V).
// ---------------------------------------------------------------------------
__global__ void __launch_bounds__(256) k_proj(const _Float16* __restrict__ sin16,
                                              const _Float16* __restrict__ wxT,
                                              const float* __restrict__ bx,
                                              _Float16* __restrict__ hn,
                                              _Float16* __restrict__ hT) {
  const int w    = blockIdx.x * 8 + (threadIdx.x >> 5);  // 0..8191
  const int lane = threadIdx.x & 31;
  const int h    = w & (NH - 1);
  const int m0   = (w >> 4) << 4;                        // row tile over B*S

  const _Float16* abase = sin16 + (size_t)m0 * DM;
  const _Float16* wbase = wxT + (size_t)h * DF * DM;

  v8f acc[4] = {};
  v16h a0, a1, b0[4], b1[4];
  a0 = ld_frag(abase, DM);
#pragma unroll
  for (int nf = 0; nf < 4; ++nf) b0[nf] = ld_frag(wbase + (size_t)(nf * 16) * DM, DM);

  for (int kk = 0; kk < DM / 32; kk += 2) {
    const int d1 = ((kk + 1) & (DM / 32 - 1)) << 5;
    gemm_stage(abase, wbase, d1, a0, b0, a1, b1, acc);
    const int d2 = ((kk + 2) & (DM / 32 - 1)) << 5;
    gemm_stage(abase, wbase, d2, a1, b1, a0, b0, acc);
  }

  // bias (D-layout: lane&15 is the N column)
#pragma unroll
  for (int nf = 0; nf < 4; ++nf) {
    float b = bx[h * DF + nf * 16 + (lane & 15)];
#pragma unroll
    for (int v = 0; v < 8; ++v) acc[nf][v] += b;
  }

  // per-row inverse L2 norm
  float inv[8];
#pragma unroll
  for (int v = 0; v < 8; ++v) {
    float s = 0.f;
#pragma unroll
    for (int nf = 0; nf < 4; ++nf) s += acc[nf][v] * acc[nf][v];
    s = redsum16(s);
    inv[v] = __frsqrt_rn(s + 1e-20f);
  }

  const int bIdx = m0 >> 11;          // / SS
  const int s0   = m0 & (SS - 1);
  const int hi   = lane >> 4;
  _Float16* hnh = hn + (size_t)(bIdx * NH + h) * SS * DF;
  _Float16* hTh = hT + (size_t)(bIdx * NH + h) * DF * SS;
#pragma unroll
  for (int nf = 0; nf < 4; ++nf) {
    const int col = nf * 16 + (lane & 15);
#pragma unroll
    for (int v = 0; v < 8; ++v) {
      const int srow = s0 + v + (hi << 3);
      const float hv = acc[nf][v];
      hnh[(size_t)srow * DF + col] = (_Float16)(hv * inv[v]);
      hTh[(size_t)col * SS + srow] = (_Float16)hv;
    }
  }
}

// ---------------------------------------------------------------------------
// Kernel 2: fused flash attention per head, ping-pong across key tiles.
// One wave: 16 queries, streams keys 32 at a time, online softmax.
// Next tile's K/V fragment loads are issued first so they hide under this
// tile's exp/reduction VALU work and LDS round-trip.
// P-tile D-layout -> A-layout fixup goes through a private 1KB LDS slab.
// ---------------------------------------------------------------------------
__device__ __forceinline__ void attn_load_tile(const _Float16* hnh,
                                               const _Float16* hTh,
                                               int t0,
                                               v16h (&kb)[4], v16h (&vf)[4]) {
  const _Float16* kbase = hnh + (size_t)t0 * DF;
  kb[0] = ld_frag(kbase, DF);
  kb[1] = ld_frag(kbase + 32, DF);
  kb[2] = ld_frag(kbase + 16 * DF, DF);
  kb[3] = ld_frag(kbase + 16 * DF + 32, DF);
#pragma unroll
  for (int nf = 0; nf < 4; ++nf)
    vf[nf] = ld_frag(hTh + (size_t)(nf * 16) * SS + t0, SS);
}

__device__ __forceinline__ void attn_step(const v16h (&qf)[2],
                                          const v16h (&kb)[4], const v16h (&vf)[4],
                                          _Float16* myp, int lane, int hi,
                                          v8f (&acc)[4], float (&m_s)[8],
                                          float (&l_s)[8]) {
  // scores (cosine similarity since hn rows are unit vectors): 16x32 tile
  v8f s0 = {}, s1 = {};
  s0 = wmma_f16(qf[0], kb[0], s0);
  s1 = wmma_f16(qf[0], kb[2], s1);
  s0 = wmma_f16(qf[1], kb[1], s0);
  s1 = wmma_f16(qf[1], kb[3], s1);

  // online softmax update
  float scl[8];
#pragma unroll
  for (int v = 0; v < 8; ++v) {
    float mx = redmax16(fmaxf(s0[v], s1[v]));
    float mn = fmaxf(m_s[v], mx);
    scl[v] = __expf(m_s[v] - mn);
    m_s[v] = mn;
  }
#pragma unroll
  for (int v = 0; v < 8; ++v) {
    float p0 = __expf(s0[v] - m_s[v]);
    float p1 = __expf(s1[v] - m_s[v]);
    l_s[v] = l_s[v] * scl[v] + redsum16(p0 + p1);
    const int addr = (v + (hi << 3)) * 32 + (lane & 15);     // row-major 16x32
    myp[addr]      = (_Float16)p0;
    myp[addr + 16] = (_Float16)p1;
  }
#pragma unroll
  for (int nf = 0; nf < 4; ++nf) {
    v8f t = acc[nf];
#pragma unroll
    for (int v = 0; v < 8; ++v) t[v] *= scl[v];
    acc[nf] = t;
  }

  // LDS ops from one wave are in-order; make the RAW explicit anyway.
  asm volatile("s_wait_dscnt 0x0" ::: "memory");
  v16h ap = ld_frag(myp, 32);                                 // P as A-operand
#pragma unroll
  for (int nf = 0; nf < 4; ++nf) acc[nf] = wmma_f16(ap, vf[nf], acc[nf]);
}

__global__ void __launch_bounds__(256) k_attn(const _Float16* __restrict__ hn,
                                              const _Float16* __restrict__ hT,
                                              _Float16* __restrict__ cat) {
  __shared__ __align__(16) _Float16 lds_p[8][16 * 32];
  const int w    = blockIdx.x * 8 + (threadIdx.x >> 5);  // 0..8191
  const int wid  = threadIdx.x >> 5;
  const int lane = threadIdx.x & 31;
  const int bh   = w >> 7;            // 0..63 = b*NH + h
  const int q0   = (w & 127) << 4;    // query tile start
  const int b    = bh >> 4;
  const int h    = bh & 15;

  const _Float16* hnh = hn + (size_t)bh * SS * DF;
  const _Float16* hTh = hT + (size_t)bh * DF * SS;
  _Float16* myp = &lds_p[wid][0];

  v16h qf[2];
  qf[0] = ld_frag(hnh + (size_t)q0 * DF + 0, DF);
  qf[1] = ld_frag(hnh + (size_t)q0 * DF + 32, DF);

  v8f acc[4] = {};
  float m_s[8], l_s[8];
#pragma unroll
  for (int v = 0; v < 8; ++v) { m_s[v] = -__builtin_inff(); l_s[v] = 0.f; }

  const int hi = lane >> 4;
  const int NT = SS / 32;             // 64 key tiles
  v16h kb0[4], vf0[4], kb1[4], vf1[4];
  attn_load_tile(hnh, hTh, 0, kb0, vf0);

  for (int kt = 0; kt < NT; kt += 2) {
    // even tile: prefetch odd tile's fragments, compute on kb0/vf0
    attn_load_tile(hnh, hTh, ((kt + 1) & (NT - 1)) << 5, kb1, vf1);
    attn_step(qf, kb0, vf0, myp, lane, hi, acc, m_s, l_s);
    // odd tile: prefetch next even tile's fragments, compute on kb1/vf1
    attn_load_tile(hnh, hTh, ((kt + 2) & (NT - 1)) << 5, kb0, vf0);
    attn_step(qf, kb1, vf1, myp, lane, hi, acc, m_s, l_s);
  }

  // normalize by softmax denominator and write concatenated heads (f16)
  float inv[8];
#pragma unroll
  for (int v = 0; v < 8; ++v) inv[v] = 1.0f / l_s[v];
#pragma unroll
  for (int nf = 0; nf < 4; ++nf) {
    const int col = h * DF + nf * 16 + (lane & 15);
#pragma unroll
    for (int v = 0; v < 8; ++v) {
      const int srow = q0 + v + (hi << 3);
      cat[(size_t)(b * SS + srow) * DM + col] = (_Float16)(acc[nf][v] * inv[v]);
    }
  }
}

// ---------------------------------------------------------------------------
// Kernel 3: output projection  out = cat @ Wp + bp   (f32 result)
// Same ping-pong pipeline as k_proj.
// ---------------------------------------------------------------------------
__global__ void __launch_bounds__(256) k_oproj(const _Float16* __restrict__ cat,
                                               const _Float16* __restrict__ wpT,
                                               const float* __restrict__ bp,
                                               float* __restrict__ out) {
  const int w    = blockIdx.x * 8 + (threadIdx.x >> 5);  // 0..8191
  const int lane = threadIdx.x & 31;
  const int m0   = (w >> 4) << 4;
  const int n0   = (w & 15) << 6;

  const _Float16* abase = cat + (size_t)m0 * DM;
  const _Float16* wbase = wpT + (size_t)n0 * DM;

  v8f acc[4] = {};
  v16h a0, a1, b0[4], b1[4];
  a0 = ld_frag(abase, DM);
#pragma unroll
  for (int nf = 0; nf < 4; ++nf) b0[nf] = ld_frag(wbase + (size_t)(nf * 16) * DM, DM);

  for (int kk = 0; kk < DM / 32; kk += 2) {
    const int d1 = ((kk + 1) & (DM / 32 - 1)) << 5;
    gemm_stage(abase, wbase, d1, a0, b0, a1, b1, acc);
    const int d2 = ((kk + 2) & (DM / 32 - 1)) << 5;
    gemm_stage(abase, wbase, d2, a1, b1, a0, b0, acc);
  }

  const int hi = lane >> 4;
#pragma unroll
  for (int nf = 0; nf < 4; ++nf) {
    const int col = n0 + nf * 16 + (lane & 15);
    const float bv = bp[col];
#pragma unroll
    for (int v = 0; v < 8; ++v)
      out[(size_t)(m0 + v + (hi << 3)) * DM + col] = acc[nf][v] + bv;
  }
}

// ---------------------------------------------------------------------------
extern "C" void kernel_launch(void* const* d_in, const int* in_sizes, int n_in,
                              void* d_out, int out_size, void* d_ws, size_t ws_size,
                              hipStream_t stream) {
  (void)in_sizes; (void)n_in; (void)out_size; (void)ws_size;
  const float* sin = (const float*)d_in[0];
  const float* Wx  = (const float*)d_in[1];
  const float* bx  = (const float*)d_in[2];
  const float* Wp  = (const float*)d_in[3];
  const float* bp  = (const float*)d_in[4];
  float* out = (float*)d_out;

  // workspace carve-out (~71 MB total)
  char* ws = (char*)d_ws;
  size_t off = 0;
  auto carve = [&](size_t bytes) {
    void* p = ws + off;
    off += (bytes + 255) & ~(size_t)255;
    return p;
  };
  _Float16* sin16 = (_Float16*)carve((size_t)BB * SS * DM * 2);
  _Float16* wxT   = (_Float16*)carve((size_t)NH * DF * DM * 2);
  _Float16* wpT   = (_Float16*)carve((size_t)DM * DM * 2);
  _Float16* hn    = (_Float16*)carve((size_t)BB * NH * SS * DF * 2);
  _Float16* hT    = (_Float16*)carve((size_t)BB * NH * SS * DF * 2);
  _Float16* cat   = (_Float16*)carve((size_t)BB * SS * DM * 2);

  k_cvt_sin<<<(BB * SS * DM) / 256, 256, 0, stream>>>(sin, sin16, BB * SS * DM);
  k_cvt_wxT<<<(NH * DM * DF) / 256, 256, 0, stream>>>(Wx, wxT);
  k_cvt_wpT<<<(DM * DM) / 256, 256, 0, stream>>>(Wp, wpT);

  k_proj <<<1024, 256, 0, stream>>>(sin16, wxT, bx, hn, hT);
  k_attn <<<1024, 256, 0, stream>>>(hn, hT, cat);
  k_oproj<<<1024, 256, 0, stream>>>(cat, wpT, bp, out);
}
